// encoder_74225624809723
// MI455X (gfx1250) — matrix operand order repaired and checked by reading, not compiled, not run on hardware
//
#include <hip/hip_runtime.h>
#include <hip/hip_bf16.h>

// ---------------------------------------------------------------------------
// MI455X (gfx1250) BiGRU encoder.
//
//  * All matmuls via v_wmma_f32_16x16x32_bf16 (fp32 accumulate).
//  * Scan kernel: wh resident in LDS (K-major per column so B-fragments are
//    2x ds_load_b128), h kept in LDS f32+bf16, rec tiles by 8 waves,
//    xp double-buffered into LDS by the Tensor Data Mover (TENSORcnt) and
//    overlapped with the recurrent WMMA.
//  * Fragment lane layouts per CDNA5 ISA 7.12.2:
//      A 16x32 bf16: lane (r=L&15,hi=L>>4) row r, K = [8hi..8hi+7] U [16+8hi..]
//      B 32x16 bf16: lane col r, K = [8hi..8hi+7] U [16+8hi..] (same as A)
//      C 16x16 f32 : vgpr v -> (M = v + 8hi, N = r)
// ---------------------------------------------------------------------------

typedef __attribute__((ext_vector_type(16))) __bf16 v16bf;
typedef __attribute__((ext_vector_type(8)))  __bf16 v8bf;
typedef __attribute__((ext_vector_type(8)))  float  v8f;

#if defined(__HIP_DEVICE_COMPILE__) && \
    __has_builtin(__builtin_amdgcn_tensor_load_to_lds) && \
    __has_builtin(__builtin_amdgcn_s_wait_tensorcnt)
#define HAVE_TDM 1
#else
#define HAVE_TDM 0
#endif

#if HAVE_TDM
typedef unsigned int u32x4 __attribute__((ext_vector_type(4)));
typedef int          i32x4 __attribute__((ext_vector_type(4)));
typedef int          i32x8 __attribute__((ext_vector_type(8)));
#endif

// Native f32->bf16 convert (hardware cvt, round-to-nearest-even).
static __device__ __forceinline__ __bf16 f2bf(float f) { return (__bf16)f; }

static __device__ __forceinline__ float sigmoidf_(float x) {
  return 1.0f / (1.0f + __expf(-x));
}

static __device__ __forceinline__ v16bf cat8(v8bf a, v8bf b) {
  return __builtin_shufflevector(a, b, 0, 1, 2, 3, 4, 5, 6, 7,
                                 8, 9, 10, 11, 12, 13, 14, 15);
}

#if HAVE_TDM
// Issue a TDM 2D tile load: numRows rows of rowLen f32 elements, global row
// stride strideElems, packed contiguously at ldsByteAddr. D# packing per
// CDNA5 ISA (group0 128b, group1 256b).
// This toolchain exposes the 6-arg builtin:
//   (u32x4 g0, i32x8 g1, i32x4 g2, i32x4 g3, i32x8 g4, i32 cpol)
static __device__ __forceinline__ void tdm_load_2d(const float* gsrc,
                                                   unsigned ldsByteAddr,
                                                   int rowLen, int numRows,
                                                   unsigned long long strideElems)
{
  const unsigned long long ga = (unsigned long long)(uintptr_t)gsrc;
  u32x4 g0;
  g0[0] = 1u;                                           // count=1 (valid D#)
  g0[1] = ldsByteAddr;                                  // lds_addr [63:32]
  g0[2] = (unsigned)(ga & 0xffffffffu);                 // global_addr lo
  g0[3] = (unsigned)((ga >> 32) & 0x1ffffffu)           // global_addr hi (57b)
        | (2u << 30);                                   // type=2 ("image")
  i32x8 g1;
  g1[0] = (int)(2u << 16);                              // data_size=2 (4 bytes)
  g1[1] = (int)(((unsigned)rowLen & 0xffffu) << 16);    // tensor_dim0 [63:48]
  g1[2] = (int)((((unsigned)rowLen >> 16) & 0xffffu)    // tensor_dim0 [79:64]
        | (((unsigned)numRows & 0xffffu) << 16));       // tensor_dim1 [95:80]
  g1[3] = (int)(((unsigned)rowLen & 0xffffu) << 16);    // tile_dim0 [127:112]
  g1[4] = (int)((unsigned)numRows & 0xffffu);           // tile_dim1 [143:128]
  g1[5] = (int)(unsigned)(strideElems & 0xffffffffu);   // dim0_stride [191:160]
  g1[6] = (int)(unsigned)((strideElems >> 32) & 0xffffu); // dim0_stride hi
  g1[7] = 0;
  i32x4 gz4 = {0, 0, 0, 0};
  i32x8 gz8 = {0, 0, 0, 0, 0, 0, 0, 0};
  __builtin_amdgcn_tensor_load_to_lds(g0, g1, gz4, gz4, gz8, 0);
}
#endif

// ---------------------------------------------------------------------------
// Embedding gather + mask.
// ---------------------------------------------------------------------------
__global__ void __launch_bounds__(128)
embed_mask_kernel(const int* __restrict__ ids, const float* __restrict__ emb,
                  float* __restrict__ dst, int ldd,
                  float* __restrict__ maskOut, int rows)
{
  const int row = blockIdx.x;
  if (row >= rows) return;
  const int id = ids[row];
  const float* __restrict__ src = emb + (size_t)id * 300;
  float* __restrict__ d = dst + (size_t)row * ldd;
  for (int c = threadIdx.x; c < 300; c += 128) d[c] = src[c];
  if (threadIdx.x == 0) maskOut[row] = (id != 0) ? 1.0f : 0.0f;
}

// ---------------------------------------------------------------------------
// Input projection GEMM: C[rows x N] = A[rows x K] * W[K x N] + bias[N].
// One wave per 16x16 tile. rows is a multiple of 16 (12800 / 1600).
// W tile staged transposed (K-major per column) so both fragments load as
// contiguous 16B LDS vectors. Edge tiles handled with wave-uniform branches
// plus clamp+select loads (no divergent load predication).
// ---------------------------------------------------------------------------
__global__ void __launch_bounds__(32)
xproj_gemm_bf16(const float* __restrict__ A, int lda, int K,
                const float* __restrict__ W, int N,
                const float* __restrict__ bias,
                float* __restrict__ C, int ldc, int rows)
{
  __shared__ __align__(16) __bf16 As[16][32];   // [row][k]
  __shared__ __align__(16) __bf16 WsT[16][32];  // [col][k]  (K-major!)

  const int lane = threadIdx.x & 31;
  const int r    = lane & 15;
  const int hi   = lane >> 4;
  const int mBase = blockIdx.x * 16;
  const int nBase = blockIdx.y * 16;

  const int bcol = nBase + r < N ? nBase + r : N - 1;
  const float bv = (nBase + r < N) ? bias[bcol] : 0.0f;
  v8f acc;
#pragma unroll
  for (int i = 0; i < 8; ++i) acc[i] = bv;

  for (int k0 = 0; k0 < K; k0 += 32) {
    const bool fullK = (k0 + 32 <= K);         // wave-uniform
    // ---- stage A tile: row (mBase+r), 16 ks starting at k0 + 16*hi ----
    {
      const int row = mBase + r;               // always < rows
      const int kk  = k0 + hi * 16;
      if (fullK) {
        const float4* __restrict__ p = (const float4*)(A + (size_t)row * lda + kk);
#pragma unroll
        for (int j = 0; j < 4; ++j) {
          float4 f = p[j];
          As[r][hi * 16 + 4 * j + 0] = f2bf(f.x);
          As[r][hi * 16 + 4 * j + 1] = f2bf(f.y);
          As[r][hi * 16 + 4 * j + 2] = f2bf(f.z);
          As[r][hi * 16 + 4 * j + 3] = f2bf(f.w);
        }
      } else {
#pragma unroll
        for (int j = 0; j < 16; ++j) {
          const int k  = kk + j;
          const int ks = k < K ? k : K - 1;    // clamp, load, select
          float f = A[(size_t)row * lda + ks];
          As[r][hi * 16 + j] = f2bf(k < K ? f : 0.0f);
        }
      }
    }
    // ---- stage W tile transposed: lane loads W row (k0+lane) ----
    {
      const int k  = k0 + lane;
      const int ks = k < K ? k : K - 1;
      if (nBase + 16 <= N) {                   // wave-uniform
        const float4* __restrict__ p = (const float4*)(W + (size_t)ks * N + nBase);
#pragma unroll
        for (int j = 0; j < 4; ++j) {
          float4 f = p[j];
          const bool ok = k < K;
          WsT[4 * j + 0][lane] = f2bf(ok ? f.x : 0.0f);
          WsT[4 * j + 1][lane] = f2bf(ok ? f.y : 0.0f);
          WsT[4 * j + 2][lane] = f2bf(ok ? f.z : 0.0f);
          WsT[4 * j + 3][lane] = f2bf(ok ? f.w : 0.0f);
        }
      } else {
#pragma unroll
        for (int j = 0; j < 16; ++j) {
          const int n  = nBase + j;
          const int ns = n < N ? n : N - 1;
          float f = W[(size_t)ks * N + ns];
          WsT[j][lane] = f2bf((k < K && n < N) ? f : 0.0f);
        }
      }
    }
    __syncthreads();

    // ---- fragments: contiguous 16B LDS vector loads ----
    const v8bf* pa = (const v8bf*)&As[r][0];
    const v8bf* pb = (const v8bf*)&WsT[r][0];
    v16bf fa = cat8(pa[hi], pa[2 + hi]);          // K 8hi..+7, 16+8hi..+7
    v16bf fb = cat8(pb[hi], pb[2 + hi]);          // K 8hi..+7, 16+8hi..+7

    acc = __builtin_amdgcn_wmma_f32_16x16x32_bf16(
        false, fa, false, fb, (short)0, acc, false, false);
    __syncthreads();
  }

  const int col = nBase + r;
  if (col < N) {
#pragma unroll
    for (int v = 0; v < 8; ++v) {
      const int m = mBase + v + 8 * hi;
      C[(size_t)m * ldc + col] = acc[v];
    }
  }
}

// ---------------------------------------------------------------------------
// GRU scan. grid.x = 2 (fwd / bwd), 256 threads (8 waves).
// xp (B,T,N3) holds x@wx + b0. Each step:
//   rec = h @ wh + b1 (WMMA, wh resident in LDS K-major)
//   z = sig(xz+rz); r = sig(xr+rr); hh = tanh(xh + r*rh)
//   h  = mask ? z*h + (1-z)*hh : h
// xp step slice TDM-double-buffered into LDS, overlapped with the WMMA.
// Dynamic LDS: whs2[Npad][128]bf16 | hbf[32][128] | hf[32][128]f32 |
//              rec[32][Npad]f32 | xbuf[2][32][N3]f32.
// ---------------------------------------------------------------------------
__global__ void __launch_bounds__(256)
gru_scan_bf16(const float* __restrict__ xpF, const float* __restrict__ xpB,
              const float* __restrict__ whF, const float* __restrict__ whB,
              const float* __restrict__ bF,  const float* __restrict__ bB,
              const int* __restrict__ ids,
              float* __restrict__ ys, int ldy,
              float* __restrict__ hOut,
              int T, int U, int N3)
{
  extern __shared__ __align__(16) char smem[];
  const int Npad = (N3 + 15) & ~15;

  __bf16* whs2 = (__bf16*)smem;                                  // [Npad][128] K-major
  __bf16* hbf  = (__bf16*)(smem + (size_t)Npad * 128 * 2);       // [32][128]
  float*  hf   = (float*)((char*)hbf + 32 * 128 * 2);            // [32][128]
  float*  rec  = (float*)((char*)hf + 32 * 128 * 4);             // [32][Npad]
  float*  xbuf = (float*)((char*)rec + (size_t)32 * Npad * 4);   // [2][32][N3]

  const int dir = blockIdx.x;
  const float* __restrict__ xp = dir ? xpB : xpF;
  const float* __restrict__ wh = dir ? whB : whF;
  const float* __restrict__ b1 = (dir ? bB : bF) + N3;           // b[1] row

  const int tid  = threadIdx.x;
  const int lane = tid & 31;
  const int wave = tid >> 5;
  const int r    = lane & 15;
  const int hi   = lane >> 4;

  // Preload wh -> bf16 LDS, K-major per column: whs2[n*128 + k]; zero h.
  for (int idx = tid; idx < Npad * 128; idx += 256) {
    const int n = idx >> 7, k = idx & 127;
    const int ks = k < U ? k : U - 1;
    const int ns = n < N3 ? n : N3 - 1;
    const float v = wh[(size_t)ks * N3 + ns];
    whs2[idx] = f2bf((k < U && n < N3) ? v : 0.0f);
  }
  for (int idx = tid; idx < 32 * 128; idx += 256) {
    hf[idx]  = 0.0f;
    hbf[idx] = f2bf(0.0f);
  }
  __syncthreads();

#if HAVE_TDM
  // Kick off TDM for step 0: 32 rows (batch) x N3, row stride T*N3.
  if (tid < 32) {
    const int ta0 = dir ? (T - 1) : 0;
    tdm_load_2d(xp + (size_t)ta0 * N3,
                (unsigned)(uintptr_t)xbuf,
                N3, 32, (unsigned long long)T * (unsigned)N3);
  }
#endif

  const int nTiles = (Npad >> 4) * 2;

  for (int t = 0; t < T; ++t) {
    const int ta = dir ? (T - 1 - t) : t;

#if HAVE_TDM
    // Issue next step's tile into the other buffer; overlaps with the WMMA.
    if (t + 1 < T && tid < 32) {
      const int tn = dir ? (T - 2 - t) : (t + 1);
      tdm_load_2d(xp + (size_t)tn * N3,
                  (unsigned)(uintptr_t)(xbuf + (size_t)((t + 1) & 1) * 32 * N3),
                  N3, 32, (unsigned long long)T * (unsigned)N3);
    }
#endif

    // ---- rec = h @ wh + b1 ----
    for (int tile = wave; tile < nTiles; tile += 8) {
      const int mTile = tile & 1;
      const int nTile = tile >> 1;
      const int col   = nTile * 16 + r;
      const int row   = mTile * 16 + r;

      const int cs = col < N3 ? col : N3 - 1;
      const float bv = (col < N3) ? b1[cs] : 0.0f;
      v8f acc;
#pragma unroll
      for (int i = 0; i < 8; ++i) acc[i] = bv;

      const v8bf* pa = (const v8bf*)(hbf + row * 128);
      const v8bf* pb = (const v8bf*)(whs2 + (size_t)col * 128);
#pragma unroll
      for (int k0 = 0; k0 < 128; k0 += 32) {
        v16bf fa = cat8(pa[(k0 >> 3) + hi], pa[(k0 >> 3) + 2 + hi]);
        v16bf fb = cat8(pb[(k0 >> 3) + hi], pb[(k0 >> 3) + 2 + hi]);
        acc = __builtin_amdgcn_wmma_f32_16x16x32_bf16(
            false, fa, false, fb, (short)0, acc, false, false);
      }
#pragma unroll
      for (int v = 0; v < 8; ++v)
        rec[(mTile * 16 + v + 8 * hi) * Npad + col] = acc[v];
    }

#if HAVE_TDM
    // TDM completes in order: <=1 outstanding means step t's tile landed.
    if (tid < 32) {
      if (t + 1 < T) __builtin_amdgcn_s_wait_tensorcnt(1);
      else           __builtin_amdgcn_s_wait_tensorcnt(0);
    }
#endif
    __syncthreads();

    // ---- fused gates (division-free indexing over padded 32x128 space) ----
#if HAVE_TDM
    const float* __restrict__ xcur = xbuf + (size_t)(t & 1) * 32 * N3;
#endif
    for (int idx = tid; idx < 32 * 128; idx += 256) {
      const int b = idx >> 7, u = idx & 127;
      if (u >= U) continue;
#if HAVE_TDM
      const float xz = xcur[b * N3 + u];
      const float xr = xcur[b * N3 + U + u];
      const float xh = xcur[b * N3 + 2 * U + u];
#else
      const size_t xrow = ((size_t)b * T + ta) * (size_t)N3;
      const float xz = xp[xrow + u];
      const float xr = xp[xrow + U + u];
      const float xh = xp[xrow + 2 * U + u];
      if (t + 1 < T) {
        const int tn = dir ? (T - 2 - t) : (t + 1);
        __builtin_prefetch(&xp[((size_t)b * T + tn) * (size_t)N3 + u], 0, 1);
      }
#endif
      const float rz = rec[b * Npad + u];
      const float rr = rec[b * Npad + U + u];
      const float rh = rec[b * Npad + 2 * U + u];

      const float z  = sigmoidf_(xz + rz);
      const float rg = sigmoidf_(xr + rr);
      const float hh = tanhf(xh + rg * rh);
      const float hp = hf[b * 128 + u];
      float hn = z * hp + (1.0f - z) * hh;
      if (ids) {
        const bool mt = ids[(size_t)b * T + ta] != 0;
        hn = mt ? hn : hp;
      }
      hf[b * 128 + u]  = hn;
      hbf[b * 128 + u] = f2bf(hn);
      ys[((size_t)b * T + ta) * (size_t)ldy + dir * U + u] = hn;
    }
    __syncthreads();
  }

  if (hOut) {
    for (int idx = tid; idx < 32 * 128; idx += 256) {
      const int b = idx >> 7, u = idx & 127;
      if (u < U) hOut[(size_t)dir * 32 * U + (size_t)b * U + u] = hf[b * 128 + u];
    }
  }
}

// ---------------------------------------------------------------------------
// Host side
// ---------------------------------------------------------------------------
extern "C" void kernel_launch(void* const* d_in, const int* in_sizes, int n_in,
                              void* d_out, int out_size, void* d_ws, size_t ws_size,
                              hipStream_t stream)
{
  (void)in_sizes; (void)n_in; (void)out_size; (void)ws_size;

  constexpr int B = 32, LC = 400, LQ = 50, DCH = 100, U = 128, CHU = 100;

  const int*   w_cont   = (const int*)d_in[0];
  const int*   w_query  = (const int*)d_in[1];
  const float* ch_cont  = (const float*)d_in[2];
  const float* ch_query = (const float*)d_in[3];

  // params flattened as a JAX pytree (sorted dict keys):
  // ch_con, ch_que, con1, con2, con3, emb, que1, que2, que3;
  // each bidir: bwd < fwd; each gru: b < wh < wx.
  auto P = [&](int i) { return (const float*)d_in[4 + i]; };
  struct Gru   { const float* b; const float* wh; const float* wx; };
  struct Bidir { Gru bwd, fwd; };
  auto getBD = [&](int base) {
    Bidir bd;
    bd.bwd = { P(base + 0), P(base + 1), P(base + 2) };
    bd.fwd = { P(base + 3), P(base + 4), P(base + 5) };
    return bd;
  };
  const Bidir ch_con = getBD(0),  ch_que = getBD(6);
  const Bidir con1   = getBD(12), con2   = getBD(18), con3 = getBD(24);
  const float* emb   = P(30);
  const Bidir que1   = getBD(31), que2   = getBD(37), que3 = getBD(43);

  // Workspace layout (floats).
  float* ws  = (float*)d_ws;
  float* ce  = ws;                           // [B*LC, 500] word(300)|char(200)
  float* qe  = ce  + (size_t)B * LC * 500;   // [B*LQ, 500]
  float* xpF = qe  + (size_t)B * LQ * 500;   // [B*LC, 384] max
  float* xpB = xpF + (size_t)B * LC * 384;
  float* y1  = xpB + (size_t)B * LC * 384;   // [B*LC, 256]
  float* y2  = y1  + (size_t)B * LC * 256;   // [B*LC, 256]

  // Output layout (floats, concatenated return tuple).
  float* out   = (float*)d_out;
  float* c3    = out;
  float* cs1   = c3  + (size_t)B * LC * 256;
  float* cs2   = cs1 + (size_t)B * U;
  float* q3    = cs2 + (size_t)B * U;
  float* qs1   = q3  + (size_t)B * LQ * 256;
  float* qs2   = qs1 + (size_t)B * U;
  float* cmask = qs2 + (size_t)B * U;
  float* qmask = cmask + (size_t)B * LC;

  auto gemm = [&](const float* A, int lda, int K,
                  const float* Wx, const float* bias, int N,
                  float* C, int ldc, int rows) {
    dim3 grid(rows / 16, (N + 15) / 16);
    xproj_gemm_bf16<<<grid, 32, 0, stream>>>(A, lda, K, Wx, N, bias, C, ldc, rows);
  };

  auto scan = [&](const float* xF, const float* xB, const Bidir& p,
                  const int* ids, float* ysOut, int ldy, float* hO,
                  int T, int Uu, int N3) {
    const int Npad = (N3 + 15) & ~15;
    const size_t sm = (size_t)Npad * 128 * 2   // wh bf16 (K-major)
                    + 32 * 128 * 2             // h bf16
                    + 32 * 128 * 4             // h f32
                    + (size_t)32 * Npad * 4    // rec f32
                    + (size_t)2 * 32 * N3 * 4; // xp double buffer
    (void)hipFuncSetAttribute(reinterpret_cast<const void*>(gru_scan_bf16),
                              hipFuncAttributeMaxDynamicSharedMemorySize, (int)sm);
    gru_scan_bf16<<<dim3(2), 256, sm, stream>>>(
        xF, xB, p.fwd.wh, p.bwd.wh, p.fwd.b, p.bwd.b,
        ids, ysOut, ldy, hO, T, Uu, N3);
  };

  // 1) word embeddings into concat buffers + masks
  embed_mask_kernel<<<B * LC, 128, 0, stream>>>(w_cont,  emb, ce, 500, cmask, B * LC);
  embed_mask_kernel<<<B * LQ, 128, 0, stream>>>(w_query, emb, qe, 500, qmask, B * LQ);

  // 2) char BiGRUs (no mask); outputs land in cols [300,500) of ce/qe
  gemm(ch_cont, DCH, DCH, ch_con.fwd.wx, ch_con.fwd.b, 3 * CHU, xpF, 3 * CHU, B * LC);
  gemm(ch_cont, DCH, DCH, ch_con.bwd.wx, ch_con.bwd.b, 3 * CHU, xpB, 3 * CHU, B * LC);
  scan(xpF, xpB, ch_con, nullptr, ce + 300, 500, nullptr, LC, CHU, 3 * CHU);

  gemm(ch_query, DCH, DCH, ch_que.fwd.wx, ch_que.fwd.b, 3 * CHU, xpF, 3 * CHU, B * LQ);
  gemm(ch_query, DCH, DCH, ch_que.bwd.wx, ch_que.bwd.b, 3 * CHU, xpB, 3 * CHU, B * LQ);
  scan(xpF, xpB, ch_que, nullptr, qe + 300, 500, nullptr, LQ, CHU, 3 * CHU);

  // 3) context word-layer stack
  gemm(ce, 500, 500, con1.fwd.wx, con1.fwd.b, 3 * U, xpF, 3 * U, B * LC);
  gemm(ce, 500, 500, con1.bwd.wx, con1.bwd.b, 3 * U, xpB, 3 * U, B * LC);
  scan(xpF, xpB, con1, w_cont, y1, 2 * U, nullptr, LC, U, 3 * U);

  gemm(y1, 2 * U, 2 * U, con2.fwd.wx, con2.fwd.b, 3 * U, xpF, 3 * U, B * LC);
  gemm(y1, 2 * U, 2 * U, con2.bwd.wx, con2.bwd.b, 3 * U, xpB, 3 * U, B * LC);
  scan(xpF, xpB, con2, w_cont, y2, 2 * U, nullptr, LC, U, 3 * U);

  gemm(y2, 2 * U, 2 * U, con3.fwd.wx, con3.fwd.b, 3 * U, xpF, 3 * U, B * LC);
  gemm(y2, 2 * U, 2 * U, con3.bwd.wx, con3.bwd.b, 3 * U, xpB, 3 * U, B * LC);
  scan(xpF, xpB, con3, w_cont, c3, 2 * U, cs1, LC, U, 3 * U);   // cs2 = cs1 + 4096

  // 4) query word-layer stack (reuse xp / y buffers)
  gemm(qe, 500, 500, que1.fwd.wx, que1.fwd.b, 3 * U, xpF, 3 * U, B * LQ);
  gemm(qe, 500, 500, que1.bwd.wx, que1.bwd.b, 3 * U, xpB, 3 * U, B * LQ);
  scan(xpF, xpB, que1, w_query, y1, 2 * U, nullptr, LQ, U, 3 * U);

  gemm(y1, 2 * U, 2 * U, que2.fwd.wx, que2.fwd.b, 3 * U, xpF, 3 * U, B * LQ);
  gemm(y1, 2 * U, 2 * U, que2.bwd.wx, que2.bwd.b, 3 * U, xpB, 3 * U, B * LQ);
  scan(xpF, xpB, que2, w_query, y2, 2 * U, nullptr, LQ, U, 3 * U);

  gemm(y2, 2 * U, 2 * U, que3.fwd.wx, que3.fwd.b, 3 * U, xpF, 3 * U, B * LQ);
  gemm(y2, 2 * U, 2 * U, que3.bwd.wx, que3.bwd.b, 3 * U, xpB, 3 * U, B * LQ);
  scan(xpF, xpB, que3, w_query, q3, 2 * U, qs1, LQ, U, 3 * U);  // qs2 = qs1 + 4096
}
